// CrissCrossAttention_52578989637701
// MI455X (gfx1250) — compile-verified, matrix operand order checked by reading, not run on hardware
//
#include <hip/hip_runtime.h>

// ---------------------------------------------------------------- types
typedef __bf16 bhalf;
typedef __attribute__((ext_vector_type(16))) bhalf v16bf;
typedef __attribute__((ext_vector_type(8)))  bhalf v8bf;
typedef __attribute__((ext_vector_type(8)))  float v8f;

#define C_DIM 512
#define HD    64
#define HH    96
#define WW    96
#define HWPIX 9216          // 96*96
#define BATCH 4
#define NBH   32            // BATCH * NUM_HEADS

// Load a 16x32 bf16 WMMA fragment (A-style or B-from-[col][k]-style) from an
// LDS tile stored row-major with `stride` bf16 elements per row.
// CDNA5 16-bit operand layout: lanes 0-15 row r hold K{0..7,16..23},
// lanes 16-31 row r hold K{8..15,24..31}; element pairs pack low/high halves.
static __device__ __forceinline__ v16bf ld_frag(const bhalf* p0, int stride) {
  const int lane = (int)(threadIdx.x & 31u);
  const bhalf* p = p0 + (lane & 15) * stride + ((lane >> 4) << 3);
  v8bf lo = *(const v8bf*)(p);
  v8bf hi = *(const v8bf*)(p + 16);
  v16bf r;
#pragma unroll
  for (int j = 0; j < 8; ++j) { r[j] = lo[j]; r[j + 8] = hi[j]; }
  return r;
}

// ---------------------------------------------------------------- 0: weights -> bf16
__global__ __launch_bounds__(256) void k_cvt_w(const float* __restrict__ Wq,
                                               const float* __restrict__ Wk,
                                               const float* __restrict__ Wv,
                                               bhalf* __restrict__ out) {
  int i = blockIdx.x * 256 + threadIdx.x;          // 262144 per matrix
  out[i]          = (bhalf)(Wq[i] * 0.125f);       // fold SCALING into Wq
  out[262144 + i] = (bhalf)Wk[i];
  out[524288 + i] = (bhalf)Wv[i];
}

// ---------------------------------------------------------------- 1: 1x1 conv GEMMs
// out[b, o, pix] = sum_c W[o,c] * x[b, c, pix]  (+ bias), written as bf16.
// Tile: 128(M=o) x 128(N=pix) per block, K step 32. 8 waves, each 32x64.
__global__ __launch_bounds__(256) void k_conv(const float* __restrict__ qin,
                                              const float* __restrict__ vin,
                                              const bhalf* __restrict__ wbf,
                                              const float* __restrict__ bq,
                                              const float* __restrict__ bk,
                                              const float* __restrict__ bv,
                                              bhalf* __restrict__ tb,
                                              bhalf* __restrict__ fb,
                                              bhalf* __restrict__ gb) {
  const int nt = blockIdx.x;                 // 0..71
  const int mt = blockIdx.y;                 // 0..3
  const int which = blockIdx.z >> 2;         // 0=t 1=f 2=g
  const int b     = blockIdx.z & 3;
  const float* x    = (which == 2) ? vin : qin;
  const bhalf* W    = wbf + (size_t)which * (C_DIM * C_DIM);
  const float* bias = (which == 0) ? bq : (which == 1) ? bk : bv;
  const float bscale = (which == 0) ? 0.125f : 1.0f;
  bhalf* out = (which == 0) ? tb : (which == 1) ? fb : gb;

  __shared__ bhalf lA[128 * 40];             // [m][k], padded stride
  __shared__ bhalf lB[128 * 40];             // [n][k] (transposed on store)

  const int tid = threadIdx.x, lane = tid & 31, wv = tid >> 5;
  const int wm = (wv >> 1) << 5;             // 0,32,64,96
  const int wn = (wv & 1) << 6;              // 0,64
  const int pix0 = nt * 128;

  v8f acc[2][4] = {};
  for (int k = 0; k < C_DIM; k += 32) {
    {   // A tile: weights (already bf16), contiguous copy
      int r = tid >> 1, c0 = (tid & 1) * 16;
      const bhalf* src = W + (size_t)(mt * 128 + r) * C_DIM + k + c0;
      bhalf* dst = &lA[r * 40 + c0];
#pragma unroll
      for (int j = 0; j < 16; ++j) dst[j] = src[j];
    }
    {   // B tile: f32 activations, convert + transpose to [pix][k]
      int c = tid >> 3, p0 = (tid & 7) * 16;
      const float* src = x + ((size_t)b * C_DIM + k + c) * HWPIX + pix0 + p0;
#pragma unroll
      for (int j = 0; j < 16; ++j) lB[(p0 + j) * 40 + c] = (bhalf)src[j];
    }
    __syncthreads();
    v16bf af[2], bfr[4];
#pragma unroll
    for (int i = 0; i < 2; ++i) af[i] = ld_frag(&lA[(wm + i * 16) * 40], 40);
#pragma unroll
    for (int j = 0; j < 4; ++j) bfr[j] = ld_frag(&lB[(wn + j * 16) * 40], 40);
#pragma unroll
    for (int i = 0; i < 2; ++i)
#pragma unroll
      for (int j = 0; j < 4; ++j)
        acc[i][j] = __builtin_amdgcn_wmma_f32_16x16x32_bf16(
            false, af[i], false, bfr[j], (short)0, acc[i][j], false, false);
    __syncthreads();
  }
#pragma unroll
  for (int i = 0; i < 2; ++i)
#pragma unroll
    for (int j = 0; j < 4; ++j) {
      const int mb = mt * 128 + wm + i * 16 + ((lane >> 4) << 3);
      const int nb = pix0 + wn + j * 16 + (lane & 15);
#pragma unroll
      for (int e = 0; e < 8; ++e) {
        const int o = mb + e;
        float valf = acc[i][j][e] + bscale * bias[o];
        out[((size_t)b * C_DIM + o) * HWPIX + nb] = (bhalf)valf;
      }
    }
}

// ---------------------------------------------------------------- 2: energies
// mode 0: e_row[bh][y=s][x][i] = sum_d t[bh,d,s,x] * f[bh,d,s,i]
// mode 1: e_col[bh][x=s][y][i] = sum_d t[bh,d,y,s] * f[bh,d,i,s]
// 96x96 output, K=64. 4 waves, each 48x48 (3x3 tiles).
__global__ __launch_bounds__(128) void k_energy(const bhalf* __restrict__ tb,
                                                const bhalf* __restrict__ fb,
                                                float* __restrict__ erow,
                                                float* __restrict__ ecol,
                                                int mode) {
  const int bh = blockIdx.x / 96, s = blockIdx.x % 96;
  __shared__ bhalf lT[96 * 72];              // [pos][d]
  __shared__ bhalf lF[96 * 72];
  const int tid = threadIdx.x, lane = tid & 31, wv = tid >> 5;
  {
    const int d = tid >> 1, h0 = (tid & 1) * 48;
    const size_t base = (size_t)(bh * HD + d) * HWPIX +
                        (mode == 0 ? (size_t)s * WW : (size_t)s);
    const int gstride = (mode == 0) ? 1 : WW;
    const bhalf* ts = tb + base;
    const bhalf* fs = fb + base;
#pragma unroll 4
    for (int j = 0; j < 48; ++j) {
      lT[(h0 + j) * 72 + d] = ts[(size_t)(h0 + j) * gstride];
      lF[(h0 + j) * 72 + d] = fs[(size_t)(h0 + j) * gstride];
    }
  }
  __syncthreads();
  const int wm = (wv >> 1) * 48, wn = (wv & 1) * 48;
  v8f acc[3][3] = {};
#pragma unroll
  for (int kc = 0; kc < 64; kc += 32) {
    v16bf af[3], bfr[3];
#pragma unroll
    for (int i = 0; i < 3; ++i) af[i] = ld_frag(&lT[(wm + i * 16) * 72 + kc], 72);
#pragma unroll
    for (int j = 0; j < 3; ++j) bfr[j] = ld_frag(&lF[(wn + j * 16) * 72 + kc], 72);
#pragma unroll
    for (int i = 0; i < 3; ++i)
#pragma unroll
      for (int j = 0; j < 3; ++j)
        acc[i][j] = __builtin_amdgcn_wmma_f32_16x16x32_bf16(
            false, af[i], false, bfr[j], (short)0, acc[i][j], false, false);
  }
  float* ebuf = (mode == 0) ? erow : ecol;
#pragma unroll
  for (int i = 0; i < 3; ++i)
#pragma unroll
    for (int j = 0; j < 3; ++j) {
      const int mb = wm + i * 16 + ((lane >> 4) << 3);
      const int nb = wn + j * 16 + (lane & 15);
#pragma unroll
      for (int e = 0; e < 8; ++e)
        ebuf[(((size_t)bh * 96 + s) * 96 + (mb + e)) * 96 + nb] = acc[i][j][e];
    }
}

// ---------------------------------------------------------------- 3: joint softmax
// One wave per (bh,y,x): 96 row logits + 96 col logits (diag masked), in place.
__global__ __launch_bounds__(256) void k_softmax(float* __restrict__ erow,
                                                 float* __restrict__ ecol) {
  const int wid  = blockIdx.x * 8 + (threadIdx.x >> 5);
  const int lane = threadIdx.x & 31;
  const int bh = wid / HWPIX;
  const int rem = wid - bh * HWPIX;
  const int y = rem / WW, x = rem - y * WW;
  float* rp = erow + (((size_t)bh * 96 + y) * 96 + x) * 96;
  float* cp = ecol + (((size_t)bh * 96 + x) * 96 + y) * 96;
  float rv[3], cv[3], mx = -1e30f;
#pragma unroll
  for (int j = 0; j < 3; ++j) {
    const int i = lane + j * 32;
    rv[j] = rp[i];
    float c = cp[i];
    cv[j] = (i == y) ? -1e30f : c;                 // diagonal mask
    mx = fmaxf(mx, fmaxf(rv[j], cv[j]));
  }
#pragma unroll
  for (int o = 16; o > 0; o >>= 1) mx = fmaxf(mx, __shfl_xor(mx, o));
  float sum = 0.f;
#pragma unroll
  for (int j = 0; j < 3; ++j) {
    rv[j] = __expf(rv[j] - mx);
    cv[j] = __expf(cv[j] - mx);
    sum += rv[j] + cv[j];
  }
#pragma unroll
  for (int o = 16; o > 0; o >>= 1) sum += __shfl_xor(sum, o);
  const float inv = __frcp_rn(sum);
#pragma unroll
  for (int j = 0; j < 3; ++j) {
    const int i = lane + j * 32;
    rp[i] = rv[j] * inv;
    cp[i] = cv[j] * inv;
  }
}

// ---------------------------------------------------------------- 4: row aggregation
// O[bh,d,y,x] = sum_i g[bh,d,y,i] * a_row[bh,y,x,i]   (64x96, K=96)
__global__ __launch_bounds__(128) void k_agg_row(const bhalf* __restrict__ gb,
                                                 const float* __restrict__ arow,
                                                 float* __restrict__ obuf) {
  const int bh = blockIdx.x / 96, y = blockIdx.x % 96;
  __shared__ bhalf lG[64 * 104];               // [d][i]
  __shared__ bhalf lA[96 * 104];               // [x][i]
  const int tid = threadIdx.x, lane = tid & 31, wv = tid >> 5;
  {
    const int d = tid >> 1, h0 = (tid & 1) * 48;
    const bhalf* src = gb + (size_t)(bh * HD + d) * HWPIX + (size_t)y * WW + h0;
#pragma unroll
    for (int j = 0; j < 48; ++j) lG[d * 104 + h0 + j] = src[j];
  }
  for (int r = tid; r < 96; r += 128) {
    const float* src = arow + (((size_t)bh * 96 + y) * 96 + r) * 96;
    for (int j = 0; j < 96; ++j) lA[r * 104 + j] = (bhalf)src[j];
  }
  __syncthreads();
  const int wm = (wv >> 1) * 32, wn = (wv & 1) * 48;
  v8f acc[2][3] = {};
#pragma unroll
  for (int kc = 0; kc < 96; kc += 32) {
    v16bf af[2], bfr[3];
#pragma unroll
    for (int i = 0; i < 2; ++i) af[i] = ld_frag(&lG[(wm + i * 16) * 104 + kc], 104);
#pragma unroll
    for (int j = 0; j < 3; ++j) bfr[j] = ld_frag(&lA[(wn + j * 16) * 104 + kc], 104);
#pragma unroll
    for (int i = 0; i < 2; ++i)
#pragma unroll
      for (int j = 0; j < 3; ++j)
        acc[i][j] = __builtin_amdgcn_wmma_f32_16x16x32_bf16(
            false, af[i], false, bfr[j], (short)0, acc[i][j], false, false);
  }
#pragma unroll
  for (int i = 0; i < 2; ++i)
#pragma unroll
    for (int j = 0; j < 3; ++j) {
      const int mb = wm + i * 16 + ((lane >> 4) << 3);
      const int nb = wn + j * 16 + (lane & 15);
#pragma unroll
      for (int e = 0; e < 8; ++e)
        obuf[(((size_t)bh * HD + (mb + e)) * 96 + y) * 96 + nb] = acc[i][j][e];
    }
}

// ---------------------------------------------------------------- 5: col aggregation + epilogue
// out[bh,d,y,x] = gamma*(O + sum_i g[bh,d,i,x]*a_col[bh,x,y,i]) + v
__global__ __launch_bounds__(128) void k_agg_col(const bhalf* __restrict__ gb,
                                                 const float* __restrict__ acol,
                                                 const float* __restrict__ obuf,
                                                 const float* __restrict__ vin,
                                                 const float* __restrict__ gamma,
                                                 float* __restrict__ outp) {
  const int bh = blockIdx.x / 96, x = blockIdx.x % 96;
  __shared__ bhalf lG[64 * 104];               // [d][i]  (g column slice)
  __shared__ bhalf lA[96 * 104];               // [y][i]
  const int tid = threadIdx.x, lane = tid & 31, wv = tid >> 5;
  {
    const int d = tid >> 1, h0 = (tid & 1) * 48;
    const bhalf* src = gb + (size_t)(bh * HD + d) * HWPIX + x;
#pragma unroll 4
    for (int j = 0; j < 48; ++j) lG[d * 104 + h0 + j] = src[(size_t)(h0 + j) * WW];
  }
  for (int r = tid; r < 96; r += 128) {
    const float* src = acol + (((size_t)bh * 96 + x) * 96 + r) * 96;
    for (int j = 0; j < 96; ++j) lA[r * 104 + j] = (bhalf)src[j];
  }
  __syncthreads();
  const float g0 = gamma[0];
  const int wm = (wv >> 1) * 32, wn = (wv & 1) * 48;
  v8f acc[2][3] = {};
#pragma unroll
  for (int kc = 0; kc < 96; kc += 32) {
    v16bf af[2], bfr[3];
#pragma unroll
    for (int i = 0; i < 2; ++i) af[i] = ld_frag(&lG[(wm + i * 16) * 104 + kc], 104);
#pragma unroll
    for (int j = 0; j < 3; ++j) bfr[j] = ld_frag(&lA[(wn + j * 16) * 104 + kc], 104);
#pragma unroll
    for (int i = 0; i < 2; ++i)
#pragma unroll
      for (int j = 0; j < 3; ++j)
        acc[i][j] = __builtin_amdgcn_wmma_f32_16x16x32_bf16(
            false, af[i], false, bfr[j], (short)0, acc[i][j], false, false);
  }
#pragma unroll
  for (int i = 0; i < 2; ++i)
#pragma unroll
    for (int j = 0; j < 3; ++j) {
      const int mb = wm + i * 16 + ((lane >> 4) << 3);
      const int nb = wn + j * 16 + (lane & 15);
#pragma unroll
      for (int e = 0; e < 8; ++e) {
        const size_t idx = (((size_t)bh * HD + (mb + e)) * 96 + nb) * 96 + x;
        outp[idx] = g0 * (obuf[idx] + acc[i][j][e]) + vin[idx];
      }
    }
}

// ---------------------------------------------------------------- launch
extern "C" void kernel_launch(void* const* d_in, const int* in_sizes, int n_in,
                              void* d_out, int out_size, void* d_ws, size_t ws_size,
                              hipStream_t stream) {
  (void)in_sizes; (void)n_in; (void)out_size; (void)ws_size;
  const float* q     = (const float*)d_in[0];
  const float* v     = (const float*)d_in[1];
  const float* Wq    = (const float*)d_in[2];
  const float* bq    = (const float*)d_in[3];
  const float* Wk    = (const float*)d_in[4];
  const float* bk    = (const float*)d_in[5];
  const float* Wv    = (const float*)d_in[6];
  const float* bv    = (const float*)d_in[7];
  const float* gamma = (const float*)d_in[8];
  float* out = (float*)d_out;

  char* ws = (char*)d_ws;
  const size_t SZ_W   = (size_t)3 * C_DIM * C_DIM * sizeof(bhalf);   // 1.5 MB
  const size_t SZ_TFG = (size_t)BATCH * C_DIM * HWPIX * sizeof(bhalf); // 36 MB each
  const size_t SZ_E   = (size_t)NBH * 96 * 96 * 96 * sizeof(float);  // 108 MB each
  bhalf* wbf  = (bhalf*)(ws);
  bhalf* tb   = (bhalf*)(ws + SZ_W);
  bhalf* fb   = (bhalf*)(ws + SZ_W + SZ_TFG);
  bhalf* gb   = (bhalf*)(ws + SZ_W + 2 * SZ_TFG);
  float* erow = (float*)(ws + SZ_W + 3 * SZ_TFG);
  float* ecol = (float*)(ws + SZ_W + 3 * SZ_TFG + SZ_E);
  float* obuf = (float*)(ws + SZ_W + 3 * SZ_TFG + 2 * SZ_E);

  k_cvt_w  <<<dim3(1024),        256, 0, stream>>>(Wq, Wk, Wv, wbf);
  k_conv   <<<dim3(72, 4, 12),   256, 0, stream>>>(q, v, wbf, bq, bk, bv, tb, fb, gb);
  k_energy <<<dim3(NBH * 96),    128, 0, stream>>>(tb, fb, erow, ecol, 0);
  k_energy <<<dim3(NBH * 96),    128, 0, stream>>>(tb, fb, erow, ecol, 1);
  k_softmax<<<dim3(36864),       256, 0, stream>>>(erow, ecol);
  k_agg_row<<<dim3(NBH * 96),    128, 0, stream>>>(gb, erow, obuf);
  k_agg_col<<<dim3(NBH * 96),    128, 0, stream>>>(gb, ecol, obuf, v, gamma, out);
}
